// FeatureLoss_45363444580426
// MI455X (gfx1250) — compile-verified, matrix operand order checked by reading, not run on hardware
//
#include <hip/hip_runtime.h>
#include <hip/hip_bf16.h>
#include <math.h>

typedef __attribute__((ext_vector_type(16))) _Float16 v16h;
typedef __attribute__((ext_vector_type(8)))  float    v8f;
typedef int i32x4 __attribute__((vector_size(16)));

union Frag16 { v16h v; unsigned u[8]; };

#define BB 16
#define DD 256
#define HH 60
#define WW 60
#define HW 3600
#define NN 1024
#define MM 2000
#define MP 2048
#define GH 240
#define GW 240

#define AS1 __attribute__((address_space(1)))
#define AS3 __attribute__((address_space(3)))

#if __has_builtin(__builtin_amdgcn_global_load_async_to_lds_b128) && \
    __has_builtin(__builtin_amdgcn_s_wait_asynccnt)
#define USE_ASYNC_LDS 1
#else
#define USE_ASYNC_LDS 0
#endif

__device__ __forceinline__ float waveSum(float v) {
    for (int off = 16; off >= 1; off >>= 1) v += __shfl_xor(v, off, 32);
    return v;
}

// ---------------------------------------------------------------- init
__global__ void init_acc(float* acc) {
    int t = threadIdx.x;
    if (t < 128) acc[t] = 0.0f;
}

// ---------------------------------------------------------------- candidates: gather + normalize -> f16, coords
__global__ __launch_bounds__(256) void prep_cand(
    const float* __restrict__ fa, const float* __restrict__ fq,
    const int* __restrict__ cand_idx,
    _Float16* __restrict__ cn_a, _Float16* __restrict__ cn_q,
    float* __restrict__ coords)
{
    int wave = (blockIdx.x * blockDim.x + threadIdx.x) >> 5;
    int lane = threadIdx.x & 31;
    int b = wave / MP, m = wave % MP;
    if (b >= BB) return;

    int idx = 0;
    float y = 1.0e9f, x = 1.0e9f;
    int inRange = (m < MM);
    if (inRange) {
        idx = cand_idx[b * MM + m];
        idx = min(max(idx, 0), HW - 1);
        y = (float)(idx / WW);
        x = (float)(idx % WW);
    }
    if (lane == 0) {
        coords[((size_t)b * MP + m) * 2 + 0] = y;
        coords[((size_t)b * MP + m) * 2 + 1] = x;
    }

    const float* srcs[2] = { fa, fq };
    _Float16*    dsts[2] = { cn_a, cn_q };
    for (int s = 0; s < 2; ++s) {
        const float* base = srcs[s] + (size_t)b * DD * HW + idx;
        float v[8], ss = 0.0f;
        #pragma unroll
        for (int j = 0; j < 8; ++j) {
            int d = lane + 32 * j;
            float t = inRange ? base[(size_t)d * HW] : 0.0f;
            v[j] = t; ss += t * t;
        }
        ss = waveSum(ss);
        float inv = 1.0f / fmaxf(sqrtf(ss), 1.0e-8f);
        _Float16* dst = dsts[s] + ((size_t)b * MP + m) * DD;
        #pragma unroll
        for (int j = 0; j < 8; ++j)
            dst[lane + 32 * j] = (_Float16)(v[j] * inv);
    }
}

// ---------------------------------------------------------------- positives: gather + normalize -> f16
__global__ __launch_bounds__(256) void prep_pos(
    const float* __restrict__ fa, const float* __restrict__ fq,
    const int* __restrict__ corrs, const int* __restrict__ valid,
    _Float16* __restrict__ pn_a, _Float16* __restrict__ pn_q)
{
    int wave = (blockIdx.x * blockDim.x + threadIdx.x) >> 5;
    int lane = threadIdx.x & 31;
    int b = wave / NN, n = wave % NN;
    if (b >= BB) return;
    float validf = (float)valid[b];

    const float* srcs[2] = { fa, fq };
    _Float16*    dsts[2] = { pn_a, pn_q };
    for (int s = 0; s < 2; ++s) {
        int yy = corrs[((b * NN + n) * 4) + 2 * s + 0];
        int xx = corrs[((b * NN + n) * 4) + 2 * s + 1];
        yy = min(max(yy, 0), HH - 1);
        xx = min(max(xx, 0), WW - 1);
        int idx = yy * WW + xx;
        const float* base = srcs[s] + (size_t)b * DD * HW + idx;
        float v[8], ss = 0.0f;
        #pragma unroll
        for (int j = 0; j < 8; ++j) {
            int d = lane + 32 * j;
            float t = base[(size_t)d * HW] * validf;
            v[j] = t; ss += t * t;
        }
        ss = waveSum(ss);
        float inv = 1.0f / fmaxf(sqrtf(ss), 1.0e-8f);
        _Float16* dst = dsts[s] + ((size_t)b * NN + n) * DD;
        #pragma unroll
        for (int j = 0; j < 8; ++j)
            dst[lane + 32 * j] = (_Float16)(v[j] * inv);
    }
}

// ---------------------------------------------------------------- WMMA hardest-negative argmin
// Block = 8 waves, all in the same batch; each wave owns 16 pos rows
// (block covers 128 rows). The 16-column candidate tile (8 KB of f16) is
// staged into LDS once per block, double-buffered; staging for tile t+1
// runs via ASYNC-to-LDS copies concurrently with the WMMAs on tile t.
__global__ __launch_bounds__(256) void argmin_wmma(
    const _Float16* __restrict__ pn,     // (B, N, D)
    const _Float16* __restrict__ cn,     // (B, MP, D)
    const int* __restrict__ corrs,       // (B, N, 4)
    int corrOff,                         // 0 for a, 2 for q
    const float* __restrict__ coords,    // (B, MP, 2)
    int* __restrict__ best)              // (B, N)
{
    __shared__ __align__(16) _Float16 Btile[2][16 * DD];   // 2 x 8KB
    __shared__ float scoords[2][32];

    int b        = blockIdx.x >> 3;          // 8 row-blocks per batch
    int rowBlock = blockIdx.x & 7;
    int wv       = threadIdx.x >> 5;         // wave in block (0..7)
    int lane     = threadIdx.x & 31;
    int t        = threadIdx.x;
    int row0     = rowBlock * 128 + wv * 16;
    int g        = lane >> 4;
    int ln       = lane & 15;

    // pos coords for the 8 rows this lane's accumulator covers (m = r + 8*g)
    float py[8], px[8];
    #pragma unroll
    for (int r = 0; r < 8; ++r) {
        int row = row0 + 8 * g + r;
        int yy = corrs[((b * NN + row) * 4) + corrOff + 0];
        int xx = corrs[((b * NN + row) * 4) + corrOff + 1];
        py[r] = (float)min(max(yy, 0), HH - 1);
        px[r] = (float)min(max(xx, 0), WW - 1);
    }

    // A fragments (ISA 7.12.2 16-bit A 16x32 layout), loaded once, reused
    // for all 128 column tiles.
    Frag16 afrag[8];
    const unsigned* aRow =
        (const unsigned*)(pn + ((size_t)b * NN + row0 + ln) * DD);
    #pragma unroll
    for (int kc = 0; kc < 8; ++kc) {
        #pragma unroll
        for (int p = 0; p < 8; ++p) {
            int k0 = (p < 4 ? 2 * p : 8 + 2 * p) + 8 * g;
            afrag[kc].u[p] = aRow[(kc * 32 + k0) >> 1];
        }
    }

    float minv[8]; int mini[8];
    #pragma unroll
    for (int r = 0; r < 8; ++r) { minv[r] = 3.0e38f; mini[r] = 0; }

    // --- tile staging: 8KB = 512 x b128; each of 256 threads moves 2 x 16B
    auto stage = [&](int ct, int bi) {
        const _Float16* src = cn + ((size_t)b * MP + ct * 16) * DD;
        #pragma unroll
        for (int i = 0; i < 2; ++i) {
            int e = (t + 256 * i) * 8;                   // half-element index
#if USE_ASYNC_LDS
            __builtin_amdgcn_global_load_async_to_lds_b128(
                (AS1 i32x4*)(unsigned long long)(const void*)(src + e),
                (AS3 i32x4*)(unsigned long long)(const void*)(&Btile[bi][e]),
                0, 0);
#else
            *(uint4*)(&Btile[bi][e]) = *(const uint4*)(src + e);
#endif
        }
        if (t < 32)
            scoords[bi][t] = coords[((size_t)b * MP + ct * 16) * 2 + t];
    };
    auto wait_stage = [&]() {
#if USE_ASYNC_LDS
        __builtin_amdgcn_s_wait_asynccnt(0);
#endif
        __syncthreads();
    };

    stage(0, 0);
    for (int ct = 0; ct < MP / 16; ++ct) {
        int bi = ct & 1;
        wait_stage();                       // tile ct resident; prev compute done
        if (ct + 1 < MP / 16) stage(ct + 1, bi ^ 1);

        int col = ct * 16 + ln;
        float cy = scoords[bi][ln * 2 + 0];
        float cx = scoords[bi][ln * 2 + 1];
        int colValid = (col < MM);

        const unsigned* bRow = (const unsigned*)(&Btile[bi][ln * DD]);
        v8f acc0 = { 0.f, 0.f, 0.f, 0.f, 0.f, 0.f, 0.f, 0.f };
        v8f acc1 = { 0.f, 0.f, 0.f, 0.f, 0.f, 0.f, 0.f, 0.f };
        #pragma unroll
        for (int kc = 0; kc < 8; kc += 2) {
            Frag16 bf0, bf1;
            #pragma unroll
            for (int p = 0; p < 8; ++p) {
                int k0 = (p < 4 ? 2 * p : 8 + 2 * p) + 8 * g;
                bf0.u[p] = bRow[((kc + 0) * 32 + k0) >> 1];
                bf1.u[p] = bRow[((kc + 1) * 32 + k0) >> 1];
            }
            acc0 = __builtin_amdgcn_wmma_f32_16x16x32_f16(
                false, afrag[kc + 0].v, false, bf0.v, (short)0, acc0, false, false);
            acc1 = __builtin_amdgcn_wmma_f32_16x16x32_f16(
                false, afrag[kc + 1].v, false, bf1.v, (short)0, acc1, false, false);
        }

        #pragma unroll
        for (int r = 0; r < 8; ++r) {
            float dot = acc0[r] + acc1[r];
            float dy = py[r] - cy, dx = px[r] - cx;
            float pix = sqrtf(dy * dy + dx * dx);
            float pen = fmaxf(7.0f - pix, 0.0f);
            float score = 1.0f - dot + 1.0e6f * pen;
            score = colValid ? score : 3.0e38f;
            if (score < minv[r]) { minv[r] = score; mini[r] = col; }
        }
    }

    // per-row argmin across the 16 lanes holding that row's columns
    #pragma unroll
    for (int off = 8; off >= 1; off >>= 1) {
        #pragma unroll
        for (int r = 0; r < 8; ++r) {
            float ov = __shfl_xor(minv[r], off, 32);
            int   oi = __shfl_xor(mini[r], off, 32);
            if (ov < minv[r]) { minv[r] = ov; mini[r] = oi; }
        }
    }
    if (ln == 0) {
        #pragma unroll
        for (int r = 0; r < 8; ++r)
            best[b * NN + row0 + 8 * g + r] = mini[r];
    }
}

// ---------------------------------------------------------------- triplet losses
__global__ __launch_bounds__(256) void loss_kernel(
    const float* __restrict__ fa, const float* __restrict__ fq,
    const int* __restrict__ corrs, const int* __restrict__ valid,
    const int* __restrict__ cand_idx,
    const int* __restrict__ best_a, const int* __restrict__ best_q,
    float* __restrict__ acc)
{
    int wave = (blockIdx.x * blockDim.x + threadIdx.x) >> 5;
    int lane = threadIdx.x & 31;
    int b = wave / NN, n = wave % NN;
    if (b >= BB) return;
    float validf = (float)valid[b];

    int ya = min(max(corrs[(b * NN + n) * 4 + 0], 0), HH - 1);
    int xa = min(max(corrs[(b * NN + n) * 4 + 1], 0), WW - 1);
    int yq = min(max(corrs[(b * NN + n) * 4 + 2], 0), HH - 1);
    int xq = min(max(corrs[(b * NN + n) * 4 + 3], 0), WW - 1);
    int iPA = ya * WW + xa;
    int iPQ = yq * WW + xq;
    int mA = min(max(best_a[b * NN + n], 0), MM - 1);
    int mQ = min(max(best_q[b * NN + n], 0), MM - 1);
    int iNA = min(max(cand_idx[b * MM + mA], 0), HW - 1);
    int iNQ = min(max(cand_idx[b * MM + mQ], 0), HW - 1);

    const float* ba = fa + (size_t)b * DD * HW;
    const float* bq = fq + (size_t)b * DD * HW;

    float d_pp = 0, s_pa = 0, s_pq = 0, d_ana = 0, s_na = 0, d_qnq = 0, s_nq = 0;
    #pragma unroll
    for (int j = 0; j < 8; ++j) {
        int d = lane + 32 * j;
        float pa = ba[(size_t)d * HW + iPA] * validf;
        float pq = bq[(size_t)d * HW + iPQ] * validf;
        float na = ba[(size_t)d * HW + iNA] * validf;
        float nq = bq[(size_t)d * HW + iNQ] * validf;
        d_pp += pa * pq;  s_pa += pa * pa;  s_pq += pq * pq;
        d_ana += pa * na; s_na += na * na;
        d_qnq += pq * nq; s_nq += nq * nq;
    }
    d_pp = waveSum(d_pp);  s_pa = waveSum(s_pa);  s_pq = waveSum(s_pq);
    d_ana = waveSum(d_ana); s_na = waveSum(s_na);
    d_qnq = waveSum(d_qnq); s_nq = waveSum(s_nq);

    if (lane == 0) {
        float npa = sqrtf(s_pa), npq = sqrtf(s_pq);
        float nna = sqrtf(s_na), nnq = sqrtf(s_nq);
        float cpos = d_pp  / fmaxf(npa * npq, 1.0e-8f);
        float cna  = d_ana / fmaxf(npa * nna, 1.0e-8f);
        float cnq  = d_qnq / fmaxf(npq * nnq, 1.0e-8f);
        float dist_pos = 0.5f * (1.0f - cpos);
        float dist_na  = 0.5f * (1.0f - cna);
        float dist_nq  = 0.5f * (1.0f - cnq);
        float pl  = fmaxf(dist_pos - 0.2f, 0.0f);
        float nla = fmaxf(0.7f - dist_na, 0.0f);
        float nlq = fmaxf(0.7f - dist_nq, 0.0f);
        const float invN = 1.0f / (float)NN;
        atomicAdd(&acc[0], validf * pl  * invN);
        atomicAdd(&acc[1], validf * nla * invN);
        atomicAdd(&acc[2], validf * nlq * invN);
    }
}

// ---------------------------------------------------------------- mask BCE + IoU
__global__ __launch_bounds__(256) void mask_kernel(
    const float* __restrict__ la, const float* __restrict__ lq,
    const int* __restrict__ ga, const int* __restrict__ gq,
    float* __restrict__ acc)
{
    int tid = blockIdx.x * blockDim.x + threadIdx.x;
    if (tid >= BB * HW) return;
    int b = tid / HW, hw = tid % HW;
    int h = hw / WW, w = hw % WW;
    int gy = h * (GH / HH), gx = w * (GW / WW);
    const float invE = 1.0f / (float)(BB * HW);

    {
        float x  = la[b * HW + hw];
        float gt = (float)ga[(b * GH + gy) * GW + gx];
        float bce = fmaxf(x, 0.0f) - x * gt + log1pf(expf(-fabsf(x)));
        atomicAdd(&acc[3], bce * invE);
        float pred = (x > 0.0f) ? 1.0f : 0.0f;
        atomicAdd(&acc[8  + b], gt * pred);
        atomicAdd(&acc[24 + b], fminf(gt + pred, 1.0f));
    }
    {
        float x  = lq[b * HW + hw];
        float gt = (float)gq[(b * GH + gy) * GW + gx];
        float bce = fmaxf(x, 0.0f) - x * gt + log1pf(expf(-fabsf(x)));
        atomicAdd(&acc[4], bce * invE);
        float pred = (x > 0.0f) ? 1.0f : 0.0f;
        atomicAdd(&acc[40 + b], gt * pred);
        atomicAdd(&acc[56 + b], fminf(gt + pred, 1.0f));
    }
}

// ---------------------------------------------------------------- finalize
__global__ void finalize_kernel(const int* __restrict__ valid,
                                const float* __restrict__ acc,
                                float* __restrict__ out)
{
    if (threadIdx.x == 0 && blockIdx.x == 0) {
        float cnt = 0.0f;
        for (int b = 0; b < BB; ++b) cnt += (float)valid[b];
        float inv = (cnt > 0.0f) ? 1.0f / fmaxf(cnt, 1.0f) : 0.0f;
        float pos_loss  = acc[0] * inv;
        float neg_loss  = 0.5f * (acc[1] * inv + acc[2] * inv);
        float mask_loss = 0.5f * (acc[3] + acc[4]);
        float ia = 0.0f, iq = 0.0f;
        for (int b = 0; b < BB; ++b) {
            ia += acc[8  + b] / fmaxf(acc[24 + b], 1.0f);
            iq += acc[40 + b] / fmaxf(acc[56 + b], 1.0f);
        }
        ia /= (float)BB; iq /= (float)BB;
        out[0] = mask_loss; out[1] = pos_loss; out[2] = neg_loss;
        out[3] = ia;        out[4] = iq;
    }
}

// ---------------------------------------------------------------- launcher
extern "C" void kernel_launch(void* const* d_in, const int* in_sizes, int n_in,
                              void* d_out, int out_size, void* d_ws, size_t ws_size,
                              hipStream_t stream) {
    const float* fa    = (const float*)d_in[0];
    const float* fq    = (const float*)d_in[1];
    const float* la    = (const float*)d_in[2];
    const float* lq    = (const float*)d_in[3];
    const int*   corrs = (const int*)d_in[4];
    const int*   valid = (const int*)d_in[5];
    const int*   cand  = (const int*)d_in[6];
    const int*   ga    = (const int*)d_in[7];
    const int*   gq    = (const int*)d_in[8];
    float*       out   = (float*)d_out;

    // workspace layout (bytes, 256-aligned)
    char* ws = (char*)d_ws;
    size_t off = 0;
    float*    acc    = (float*)(ws + off);          off += 512;
    _Float16* pn_a   = (_Float16*)(ws + off);       off += (size_t)BB * NN * DD * 2;
    _Float16* pn_q   = (_Float16*)(ws + off);       off += (size_t)BB * NN * DD * 2;
    _Float16* cn_a   = (_Float16*)(ws + off);       off += (size_t)BB * MP * DD * 2;
    _Float16* cn_q   = (_Float16*)(ws + off);       off += (size_t)BB * MP * DD * 2;
    float*    coords = (float*)(ws + off);          off += (size_t)BB * MP * 2 * 4;
    int*      best_a = (int*)(ws + off);            off += (size_t)BB * NN * 4;
    int*      best_q = (int*)(ws + off);            off += (size_t)BB * NN * 4;
    (void)off; (void)ws_size; (void)in_sizes; (void)n_in; (void)out_size;

    init_acc<<<1, 128, 0, stream>>>(acc);

    // one wave per row, 8 waves per block
    prep_cand<<<(BB * MP) / 8, 256, 0, stream>>>(fa, fq, cand, cn_a, cn_q, coords);
    prep_pos <<<(BB * NN) / 8, 256, 0, stream>>>(fa, fq, corrs, valid, pn_a, pn_q);

    // 8 waves per block, 128 rows per block: B * (N/128) = 128 blocks
    argmin_wmma<<<BB * (NN / 128), 256, 0, stream>>>(pn_a, cn_a, corrs, 0, coords, best_a);
    argmin_wmma<<<BB * (NN / 128), 256, 0, stream>>>(pn_q, cn_q, corrs, 2, coords, best_q);

    loss_kernel<<<(BB * NN) / 8, 256, 0, stream>>>(fa, fq, corrs, valid, cand,
                                                   best_a, best_q, acc);
    mask_kernel<<<(BB * HW + 255) / 256, 256, 0, stream>>>(la, lq, ga, gq, acc);
    finalize_kernel<<<1, 32, 0, stream>>>(valid, acc, out);
}
